// LSGPRM_6519760355647
// MI455X (gfx1250) — matrix operand order repaired and checked by reading, not compiled, not run on hardware
//
#include <hip/hip_runtime.h>
#include <hip/hip_bf16.h>

typedef __attribute__((ext_vector_type(16))) __bf16          v16bf;
typedef __attribute__((ext_vector_type(8)))  float           v8f;
typedef __attribute__((ext_vector_type(8)))  unsigned short  us8;
typedef __attribute__((ext_vector_type(4)))  unsigned int    u32x4;
typedef __attribute__((ext_vector_type(4)))  int             i32x4;
typedef __attribute__((ext_vector_type(8)))  int             i32x8;

#define NNODES 100000
#define HDIM   64
#define CDIM   40
#define KPROP  10
#define LDST   72   // padded n-major LDS stride (ushorts) to spread banks

#if defined(__has_builtin)
#if __has_builtin(__builtin_amdgcn_tensor_load_to_lds) && __has_builtin(__builtin_amdgcn_s_wait_tensorcnt)
#define HAVE_TDM 1
#endif
#endif

static __device__ __forceinline__ unsigned short f2bf(float f) {
    // round-to-nearest-even fp32 -> bf16
    unsigned u = __float_as_uint(f);
    unsigned r = ((u >> 16) & 1u) + 0x7FFFu;
    return (unsigned short)((u + r) >> 16);
}

__global__ void k_zero_f32(float* p, int n) {
    int i = blockIdx.x * blockDim.x + threadIdx.x;
    if (i < n) p[i] = 0.0f;
}

__global__ void k_deg(const int* __restrict__ col, float* __restrict__ deg, int E) {
    int e = blockIdx.x * blockDim.x + threadIdx.x;
    if (e < E) atomicAdd(&deg[col[e]], 1.0f);
}

__global__ void k_dinv(float* deg, int n) {
    int i = blockIdx.x * blockDim.x + threadIdx.x;
    if (i < n) deg[i] = rsqrtf(deg[i]);   // deg >= 1 (self loops)
}

__global__ void k_val(const int* __restrict__ row, const int* __restrict__ col,
                      const float* __restrict__ dinv, float* __restrict__ val, int E) {
    int e = blockIdx.x * blockDim.x + threadIdx.x;
    if (e < E) val[e] = dinv[row[e]] * dinv[col[e]];
}

// row[] sorted ascending (np.unique(row*N+col)). row_ptr[r] = lower_bound(row, r).
__global__ void k_rowptr(const int* __restrict__ row, int* __restrict__ rp, int E, int n) {
    int r = blockIdx.x * blockDim.x + threadIdx.x;
    if (r > n) return;
    int lo = 0, hi = E;
    while (lo < hi) {
        int mid = (lo + hi) >> 1;
        if (row[mid] < r) lo = mid + 1; else hi = mid;
    }
    rp[r] = lo;
}

__global__ void k_f32_to_bf16(const float* __restrict__ in, unsigned short* __restrict__ out, int n) {
    int i = blockIdx.x * blockDim.x + threadIdx.x;
    if (i < n) out[i] = f2bf(in[i]);
}

// W (rows x src_cols, row-major fp32) -> bf16 (rows x ldb), zero-padded columns
__global__ void k_w_to_bf16(const float* __restrict__ W, unsigned short* __restrict__ out,
                            int rows, int src_cols, int ldb) {
    int i = blockIdx.x * blockDim.x + threadIdx.x;
    if (i >= rows * ldb) return;
    int r = i / ldb, c = i % ldb;
    out[i] = (c < src_cols) ? f2bf(W[r * src_cols + c]) : (unsigned short)0;
}

__global__ void k_relu(const float* __restrict__ in, float* __restrict__ out, int n) {
    int i = blockIdx.x * blockDim.x + threadIdx.x;
    if (i < n) out[i] = fmaxf(in[i], 0.0f);
}

// One wave32 per 16x16 output tile. A: M x 64 bf16 row-major (global).
// B: 64 x LDB bf16 row-major -> staged into LDS (TDM for LDB==64, async copy
// for LDB==48), then transposed in-LDS to n-major (stride LDST) so B fragments
// are two ds_load_b128 per K-step.
template <int LDB>
__global__ void __launch_bounds__(256)
k_gemm_wmma(const unsigned short* __restrict__ A, const unsigned short* __restrict__ B,
            const float* __restrict__ bias, float* __restrict__ out,
            float* __restrict__ agg, const float* __restrict__ att0,
            int m_tiles, int n_tiles, int ncols, int out_stride) {
    __shared__ unsigned short sB[64 * 64];      // k-major staging target
    __shared__ unsigned short sBT[64 * LDST];   // n-major, padded

    const int tid  = threadIdx.x;
    const int wid  = (blockIdx.x * blockDim.x + tid) >> 5;
    const int lane = tid & 31;
    constexpr int NEL = 64 * LDB;               // ushorts in W tile

#if defined(HAVE_TDM)
    if (LDB == 64) {
        // --- stage W via Tensor Data Mover: one 2D descriptor, issued by wave 0 ---
        if (tid < 32) {
            unsigned ldsa = (unsigned)(size_t)(&sB[0]);
            unsigned long long ga = (unsigned long long)(size_t)B;
            u32x4 g0;
            g0.x = 1u;                                   // count=1, user mode
            g0.y = ldsa;                                 // lds_addr
            g0.z = (unsigned)(ga & 0xffffffffull);       // global_addr[31:0]
            g0.w = (unsigned)((ga >> 32) & 0x01ffffffull) | (2u << 30); // addr[56:32] | type=2
            i32x8 g1;
            g1[0] = (1 << 16);                           // wg_mask=0, data_size=1 (2B)
            g1[1] = (64 << 16);                          // tensor_dim0 = 64
            g1[2] = (64 << 16);                          // tensor_dim1 = 64
            g1[3] = (64 << 16);                          // tile_dim0 = 64
            g1[4] = 64;                                  // tile_dim1 = 64, tile_dim2 = 0
            g1[5] = 64;                                  // tensor_dim0_stride = 64
            g1[6] = (int)((4096u & 0xffffu) << 16);      // tensor_dim1_stride = 4096
            g1[7] = 0;
            i32x4 gz = {0, 0, 0, 0};
#if __clang_major__ >= 23
            i32x8 gz8 = {0, 0, 0, 0, 0, 0, 0, 0};
            __builtin_amdgcn_tensor_load_to_lds(g0, g1, gz, gz, gz8, 0);
#else
            __builtin_amdgcn_tensor_load_to_lds(g0, g1, gz, gz, 0);
#endif
            __builtin_amdgcn_s_wait_tensorcnt(0);
        }
    } else
#endif
    {
        // --- stage W: global -> LDS with CDNA5 async copy (ASYNCcnt tracked) ---
        const int nchunk = NEL / 8;              // 16B chunks
        for (int i = tid; i < nchunk; i += 256) {
            unsigned ldsa = (unsigned)(size_t)(&sB[0]) + (unsigned)(i * 16);
            unsigned long long ga = (unsigned long long)(size_t)B + (unsigned long long)(i * 16);
            asm volatile("global_load_async_to_lds_b128 %0, %1, off"
                         :: "v"(ldsa), "v"(ga) : "memory");
        }
        asm volatile("s_wait_asynccnt 0x0" ::: "memory");
    }
    __syncthreads();
    // --- in-LDS transpose: sBT[n*LDST + k] = sB[k*LDB + n] (LDB is constexpr) ---
    for (int i = tid; i < NEL; i += 256) {
        int k = i / LDB, n = i % LDB;
        sBT[n * LDST + k] = sB[i];
    }
    __syncthreads();

    if (wid < m_tiles * n_tiles) {              // wave-uniform: EXEC stays all-ones
        const int tm  = wid / n_tiles;
        const int tn  = wid % n_tiles;
        const int l16 = lane & 15;              // M for A-frag, N for B/C frags
        const int h   = lane >> 4;              // lane half

        v8f acc = {};
        const unsigned short* arow = A + (size_t)(tm * 16 + l16) * HDIM;
        const int ncol = tn * 16 + l16;
        const unsigned short* bcol = &sBT[ncol * LDST];

        #pragma unroll
        for (int k0 = 0; k0 < HDIM; k0 += 32) {
            // A fragment: 16-bit 16x32 layout -> two contiguous 16B global loads
            union { v16bf v; us8 u[2]; } af;
            af.u[0] = *(const us8*)(arow + k0 + 8 * h);
            af.u[1] = *(const us8*)(arow + k0 + 16 + 8 * h);
            // B fragment: lane holds column ncol, K = k0 + 8*h + [0..7], k0 + 16 + 8*h + [0..7]
            union { v16bf v; us8 u[2]; } bfr;
            const int kb = k0 + 8 * h;
            bfr.u[0] = *(const us8*)(bcol + kb);
            bfr.u[1] = *(const us8*)(bcol + kb + 16);
            acc = __builtin_amdgcn_wmma_f32_16x16x32_bf16(
                      false, af.v, false, bfr.v, (short)0, acc, false, false);
        }

        if (ncol < ncols) {
            float bb = bias[ncol];
            float a0 = (agg != nullptr) ? att0[0] : 0.0f;
            #pragma unroll
            for (int v = 0; v < 8; ++v) {       // C/D layout: M = v + 8*h, N = l16
                int r = tm * 16 + 8 * h + v;
                float xv = acc[v] + bb;
                out[(size_t)r * out_stride + ncol] = xv;
                if (agg) agg[(size_t)r * HDIM + ncol] = a0 * xv;
            }
        }
    }
}

// One wave32 per row (edges sorted by row -> atomics-free deterministic segment sum).
// Each lane owns 2 features. Fused: agg[row] += att[att_idx] * xout[row].
__global__ void __launch_bounds__(256)
k_spmm(const float* __restrict__ xin, float* __restrict__ xout,
       float* __restrict__ agg, const int* __restrict__ col,
       const float* __restrict__ val, const int* __restrict__ rp,
       const float* __restrict__ att, int att_idx, int nrows) {
    int wid  = (blockIdx.x * blockDim.x + threadIdx.x) >> 5;
    int lane = threadIdx.x & 31;
    if (wid >= nrows) return;
    int beg = rp[wid], end = rp[wid + 1];
    float ax = 0.0f, ay = 0.0f;
    for (int e = beg; e < end; ++e) {
        int   c = col[e];
        float v = val[e];
        if (e + 1 < end)  // prefetch next gathered row (global_prefetch_b8)
            __builtin_prefetch(xin + (size_t)col[e + 1] * HDIM + lane * 2, 0, 3);
        const float2 xc = *(const float2*)(xin + (size_t)c * HDIM + lane * 2);
        ax = fmaf(v, xc.x, ax);
        ay = fmaf(v, xc.y, ay);
    }
    float coef = att[att_idx];
    size_t o = (size_t)wid * HDIM + lane * 2;
    xout[o]     = ax;
    xout[o + 1] = ay;
    agg[o]     += coef * ax;
    agg[o + 1] += coef * ay;
}

__global__ void k_logsoftmax(float* __restrict__ out, int nrows, int ncols) {
    int r = blockIdx.x * blockDim.x + threadIdx.x;
    if (r >= nrows) return;
    float* p = out + (size_t)r * ncols;
    float m = -INFINITY;
    for (int c = 0; c < ncols; ++c) m = fmaxf(m, p[c]);
    float s = 0.0f;
    for (int c = 0; c < ncols; ++c) s += expf(p[c] - m);
    float ls = logf(s);
    for (int c = 0; c < ncols; ++c) p[c] = p[c] - m - ls;
}

extern "C" void kernel_launch(void* const* d_in, const int* in_sizes, int n_in,
                              void* d_out, int out_size, void* d_ws, size_t ws_size,
                              hipStream_t stream) {
    (void)n_in; (void)out_size; (void)ws_size;
    const float* x    = (const float*)d_in[0];
    const float* att  = (const float*)d_in[1];
    const float* W0   = (const float*)d_in[2];
    const float* b0   = (const float*)d_in[3];
    const float* W1   = (const float*)d_in[4];
    const float* b1   = (const float*)d_in[5];
    const float* Wout = (const float*)d_in[6];
    const float* bout = (const float*)d_in[7];
    const int*   ei   = (const int*)d_in[8];

    const int E  = in_sizes[8] / 2;
    const int Nn = NNODES;
    const int* row = ei;
    const int* colv = ei + E;

    // carve workspace (256B aligned)
    char* ws = (char*)d_ws;
    size_t off = 0;
    auto carve = [&](size_t bytes) -> void* {
        void* p = ws + off;
        off = (off + bytes + 255) & ~(size_t)255;
        return p;
    };
    float*          deg = (float*)carve((size_t)Nn * sizeof(float));
    float*          val = (float*)carve((size_t)E * sizeof(float));
    int*            rp  = (int*)carve((size_t)(Nn + 1) * sizeof(int));
    float*          xa  = (float*)carve((size_t)Nn * HDIM * sizeof(float));
    float*          xb  = (float*)carve((size_t)Nn * HDIM * sizeof(float));
    float*          agg = (float*)carve((size_t)Nn * HDIM * sizeof(float));
    unsigned short* xbf = (unsigned short*)carve((size_t)Nn * HDIM * 2);
    unsigned short* wbf = (unsigned short*)carve((size_t)64 * 64 * 2);

    const int T = 256;
    auto gr = [](long long n, int t) { return (int)((n + t - 1) / t); };

    // --- graph normalization ---
    k_zero_f32<<<gr(Nn, T), T, 0, stream>>>(deg, Nn);
    k_deg   <<<gr(E, T), T, 0, stream>>>(colv, deg, E);
    k_dinv  <<<gr(Nn, T), T, 0, stream>>>(deg, Nn);
    k_val   <<<gr(E, T), T, 0, stream>>>(row, colv, deg, val, E);
    k_rowptr<<<gr(Nn + 1, T), T, 0, stream>>>(row, rp, E, Nn);

    const int m_tiles = Nn / 16;               // 6250 exact
    const float* Ws[2] = {W0, W1};
    const float* bs[2] = {b0, b1};
    const float* xin = x;

    for (int L = 0; L < 2; ++L) {
        k_f32_to_bf16<<<gr((long long)Nn * HDIM, T), T, 0, stream>>>(xin, xbf, Nn * HDIM);
        k_w_to_bf16  <<<gr(64 * 64, T), T, 0, stream>>>(Ws[L], wbf, 64, 64, 64);
        int n_tiles = HDIM / 16;               // 4
        long long waves = (long long)m_tiles * n_tiles;
        k_gemm_wmma<64><<<gr(waves * 32, T), T, 0, stream>>>(
            xbf, wbf, bs[L], xa, agg, att + L * (KPROP + 1),
            m_tiles, n_tiles, HDIM, HDIM);

        float* cur = xa; float* oth = xb;
        for (int j = 1; j <= KPROP; ++j) {
            k_spmm<<<gr((long long)Nn * 32, T), T, 0, stream>>>(
                cur, oth, agg, colv, val, rp, att, L * (KPROP + 1) + j, Nn);
            float* t = cur; cur = oth; oth = t;
        }
        k_relu<<<gr((long long)Nn * HDIM, T), T, 0, stream>>>(agg, xa, Nn * HDIM);
        xin = xa;
    }

    // --- output projection: 64 -> 40 (pad B to 48 cols, guard stores) ---
    k_f32_to_bf16<<<gr((long long)Nn * HDIM, T), T, 0, stream>>>(xa, xbf, Nn * HDIM);
    k_w_to_bf16  <<<gr(64 * 48, T), T, 0, stream>>>(Wout, wbf, 64, CDIM, 48);
    {
        int n_tiles = 3;
        long long waves = (long long)m_tiles * n_tiles;
        k_gemm_wmma<48><<<gr(waves * 32, T), T, 0, stream>>>(
            xbf, wbf, bout, (float*)d_out, nullptr, nullptr,
            m_tiles, n_tiles, CDIM, CDIM);
    }
    k_logsoftmax<<<gr(Nn, T), T, 0, stream>>>((float*)d_out, Nn, CDIM);
}
